// FMGenDecoder_10634339025015
// MI455X (gfx1250) — matrix operand order repaired and checked by reading, not compiled, not run on hardware
//
#include <hip/hip_runtime.h>
#include <hip/hip_bf16.h>

// ---------------------------------------------------------------------------
// FMGenDecoder for MI455X (gfx1250, wave32).
// Memory-bound (~560 MB streamed -> ~24us floor @ 23.3 TB/s). GEMMs on
// v_wmma_f32_16x16x32_f16 (f32 accumulate); U/X tiles staged to LDS via the
// gfx1250 async global->LDS path (ASYNCcnt) with double buffering so WMMA
// overlaps the next tile's HBM fetch. feast softmax over 1 head == 1.0, so
// feast_conv == mean-aggregate of (x@W) + bias.
// ---------------------------------------------------------------------------

typedef __attribute__((ext_vector_type(16))) _Float16 v16h;
typedef __attribute__((ext_vector_type(8)))  float    v8f;
typedef __attribute__((ext_vector_type(4)))  float    f32x4;
typedef int v4i_vec __attribute__((vector_size(16)));   // b128 payload type

#define THREADS 256

#if defined(__has_builtin)
#if __has_builtin(__builtin_amdgcn_global_load_async_to_lds_b128) && \
    __has_builtin(__builtin_amdgcn_s_wait_asynccnt)
#define USE_ASYNC_LDS 1
#endif
#endif

// ---- async global->LDS b128 (ASYNCcnt) with sync fallback -----------------
__device__ __forceinline__ void stage_b128(const float* g, float* l) {
#ifdef USE_ASYNC_LDS
  __builtin_amdgcn_global_load_async_to_lds_b128(
      (__attribute__((address_space(1))) v4i_vec*)g,
      (__attribute__((address_space(3))) v4i_vec*)l, 0, 0);
#else
  *(f32x4*)l = __builtin_nontemporal_load((const f32x4*)g);
#endif
}
__device__ __forceinline__ void stage_wait5() {
#ifdef USE_ASYNC_LDS
  __builtin_amdgcn_s_wait_asynccnt(5);   // oldest stage (5 ops) retired
#endif
}
__device__ __forceinline__ void stage_wait0() {
#ifdef USE_ASYNC_LDS
  __builtin_amdgcn_s_wait_asynccnt(0);
#endif
}

// ---------------------------------------------------------------------------
// x = z @ lin_w + lin_b  (1x128 @ 128x160), then broadcast x[:32] to 64 rows.
// ---------------------------------------------------------------------------
__global__ __launch_bounds__(THREADS)
void k_lin(const float* __restrict__ z, const float* __restrict__ lw,
           const float* __restrict__ lb, float* __restrict__ xbuf,
           float* __restrict__ xg0) {
  __shared__ float zs[128];
  __shared__ float xs[160];
  int t = threadIdx.x;
  if (t < 128) zs[t] = z[t];
  __syncthreads();
  if (t < 160) {
    float acc = lb[t];
#pragma unroll 4
    for (int k = 0; k < 128; ++k) acc = fmaf(zs[k], lw[k * 160 + t], acc);
    xs[t]   = acc;
    xbuf[t] = acc;
  }
  __syncthreads();
  for (int idx = t; idx < 64 * 32; idx += THREADS) xg0[idx] = xs[idx & 31];
}

// ---------------------------------------------------------------------------
// xl = x[32:160] @ loc_w + loc_b  (128 x 524288). Stream-once 268 MB weight:
// coalesced nontemporal b128 so it does not displace L2.
// ---------------------------------------------------------------------------
__global__ __launch_bounds__(THREADS)
void k_loc(const float* __restrict__ xbuf, const float* __restrict__ lw,
           const float* __restrict__ lb, float* __restrict__ xl) {
  __shared__ float xs[128];
  int t = threadIdx.x;
  if (t < 128) xs[t] = xbuf[32 + t];
  __syncthreads();
  size_t j4 = (size_t)blockIdx.x * THREADS + t;        // one float4 per thread
  const f32x4* W4 = (const f32x4*)lw;
  f32x4 acc = ((const f32x4*)lb)[j4];
  const size_t ld4 = 524288 / 4;
#pragma unroll 4
  for (int k = 0; k < 128; ++k) {
    f32x4 w = __builtin_nontemporal_load(&W4[(size_t)k * ld4 + j4]);
    acc.x = fmaf(xs[k], w.x, acc.x);
    acc.y = fmaf(xs[k], w.y, acc.y);
    acc.z = fmaf(xs[k], w.z, acc.z);
    acc.w = fmaf(xs[k], w.w, acc.w);
  }
  ((f32x4*)xl)[j4] = acc;
}

// ---------------------------------------------------------------------------
// WMMA GEMM: D(N x 32) = U(N x M) @ X(M x 32), f32 in/out, f16 compute.
// 8 waves/block, wave owns a 16-row tile; 2 wmma per 32-wide k-chunk.
// Double-buffered async LDS staging: 5 async b128 per thread per stage.
// N % 128 == 0, M % 32 == 0 at every call site.
// ---------------------------------------------------------------------------
__global__ __launch_bounds__(THREADS)
void k_gemm_wmma(const float* __restrict__ U, const float* __restrict__ X,
                 float* __restrict__ D, int N, int M) {
  __shared__ float ldsX[2][32 * 32];        // shared 32x32 chunk of X
  __shared__ float ldsU[2][8 * 16 * 32];    // per-wave 16x32 tile of U
  const int tid     = threadIdx.x;
  const int lane    = tid & 31;
  const int wave    = tid >> 5;
  const int rowBase = blockIdx.x * 128 + wave * 16;
  const int nk      = M >> 5;

  auto issue = [&](int kc, int p) {
    // X chunk: 256 x b128 (one per thread)
    stage_b128(X + (size_t)kc * 32 + tid * 4, &ldsX[p][tid * 4]);
    // U tile: 4 x b128 per lane (16 rows x 8 float4)
    const float* Ut  = U + (size_t)rowBase * M + kc;
    float*       dst = &ldsU[p][wave * 16 * 32];
    if (kc + 32 < M) __builtin_prefetch(Ut + 32, 0, 1);  // global_prefetch_b8
#pragma unroll
    for (int r = 0; r < 4; ++r) {
      int idx  = lane + r * 32;      // 0..127
      int row  = idx >> 3;
      int col4 = idx & 7;
      stage_b128(Ut + (size_t)row * M + col4 * 4, dst + idx * 4);
    }
  };

  v8f c0 = {};
  v8f c1 = {};
  int p = 0;
  issue(0, 0);

  for (int k = 0; k < nk; ++k) {
    if (k + 1 < nk) { issue((k + 1) * 32, p ^ 1); stage_wait5(); }
    else            { stage_wait0(); }
    __syncthreads();     // stage-k LDS data visible to the whole block

    // ---- A (16x32 f16). lanes 0-15 -> K {0..15}; 16-31 -> {16..31}
    v16h a;
    {
      const float* ut  = &ldsU[p][wave * 16 * 32 + (lane & 15) * 32];
      const int    klo = (lane >> 4) * 16;
#pragma unroll
      for (int e = 0; e < 8; ++e) a[e]     = (_Float16)ut[klo + e];
#pragma unroll
      for (int e = 0; e < 8; ++e) a[8 + e] = (_Float16)ut[klo + 8 + e];
    }
    // ---- B halves (32x16 f16). lane = column, lane group selects K half.
    v16h b0, b1;
    {
      const int krow = (lane >> 4) * 16;
      const int col  = lane & 15;
#pragma unroll
      for (int e = 0; e < 16; ++e) {
        b0[e] = (_Float16)ldsX[p][(krow + e) * 32 + col];
        b1[e] = (_Float16)ldsX[p][(krow + e) * 32 + col + 16];
      }
    }

    c0 = __builtin_amdgcn_wmma_f32_16x16x32_f16(false, a, false, b0,
                                                (short)0, c0, false, false);
    c1 = __builtin_amdgcn_wmma_f32_16x16x32_f16(false, a, false, b1,
                                                (short)0, c1, false, false);
    __syncthreads();     // done reading buf p before it is rewritten
    p ^= 1;
  }

  // ---- store D: C layout -> VGPR r, lanes 0-15: row r; lanes 16-31: row r+8.
  {
    const int rowOff = (lane >> 4) * 8;
    const int col    = lane & 15;
    float* Drow = D + (size_t)rowBase * 32;
#pragma unroll
    for (int r = 0; r < 8; ++r) {
      Drow[(rowOff + r) * 32 + col]      = c0[r];
      Drow[(rowOff + r) * 32 + col + 16] = c1[r];
    }
  }
}

// ---------------------------------------------------------------------------
// xw3 = X(n x 32) @ W3(32 x 3)   (final skinny feast weight)
// ---------------------------------------------------------------------------
__global__ __launch_bounds__(THREADS)
void k_xw3(const float* __restrict__ X, const float* __restrict__ W3,
           float* __restrict__ xw3, int n) {
  __shared__ float w[96];
  int t = threadIdx.x;
  if (t < 96) w[t] = W3[t];
  __syncthreads();
  int i = blockIdx.x * THREADS + t;
  if (i >= n) return;
  const float* xr = X + (size_t)i * 32;
  float a0 = 0.f, a1 = 0.f, a2 = 0.f;
#pragma unroll
  for (int k = 0; k < 32; ++k) {
    float v = xr[k];
    a0 = fmaf(v, w[k * 3 + 0], a0);
    a1 = fmaf(v, w[k * 3 + 1], a1);
    a2 = fmaf(v, w[k * 3 + 2], a2);
  }
  xw3[(size_t)i * 3 + 0] = a0;
  xw3[(size_t)i * 3 + 1] = a1;
  xw3[(size_t)i * 3 + 2] = a2;
}

__global__ __launch_bounds__(THREADS)
void k_zero(float* __restrict__ p, int n) {
  int i = blockIdx.x * THREADS + threadIdx.x;
  if (i < n) p[i] = 0.f;
}

// ---------------------------------------------------------------------------
// Edge scatter: agg[dst] += xw[src]; cnt[dst] += 1.  e >= E -> self loop.
// edges layout: [src(E), dst(E)] (row-major (2,E) int32).
// ---------------------------------------------------------------------------
__global__ __launch_bounds__(THREADS)
void k_scatter(const int* __restrict__ edges, int E, int n,
               const float* __restrict__ xw, float* __restrict__ agg,
               float* __restrict__ cnt, int C) {
  int e = blockIdx.x * THREADS + threadIdx.x;
  if (e >= E + n) return;
  int src, dst;
  if (e < E) { src = edges[e]; dst = edges[E + e]; }
  else       { src = dst = e - E; }
  atomicAdd(&cnt[dst], 1.0f);
  const float* m = xw + (size_t)src * C;
  float*       a = agg + (size_t)dst * C;
  for (int c = 0; c < C; ++c) atomicAdd(&a[c], m[c]);
}

__global__ __launch_bounds__(THREADS)
void k_feast_fin(const float* __restrict__ agg, const float* __restrict__ cnt,
                 const float* __restrict__ b, float* __restrict__ out,
                 int n, int C) {
  int idx = blockIdx.x * THREADS + threadIdx.x;
  if (idx >= n * C) return;
  int i = idx / C, c = idx - i * C;
  float d = cnt[i];
  out[idx] = agg[idx] / fmaxf(d, 1.0f) + b[c];
}

// ---------------------------------------------------------------------------
// Batch-norm: pass 1 accumulates per-channel sum / sumsq (lane c == channel),
// pass 2 normalizes + leaky_relu(0.01) in place.
// ---------------------------------------------------------------------------
__global__ __launch_bounds__(THREADS)
void k_bn_stats(const float* __restrict__ X, float* __restrict__ stats, int n) {
  int c = threadIdx.x & 31;
  float s = 0.f, q = 0.f;
  int total = n * 32;
  for (int idx = blockIdx.x * THREADS + threadIdx.x; idx < total;
       idx += gridDim.x * THREADS) {
    float v = X[idx];
    s += v;
    q = fmaf(v, v, q);
  }
  atomicAdd(&stats[c], s);
  atomicAdd(&stats[32 + c], q);
}

__global__ __launch_bounds__(THREADS)
void k_bn_apply(float* __restrict__ X, const float* __restrict__ stats,
                const float* __restrict__ gamma, const float* __restrict__ beta,
                int n) {
  int idx = blockIdx.x * THREADS + threadIdx.x;
  if (idx >= n * 32) return;
  int c = idx & 31;
  float inv_n = 1.0f / (float)n;
  float mu  = stats[c] * inv_n;
  float var = stats[32 + c] * inv_n - mu * mu;
  float y = (X[idx] - mu) * rsqrtf(var + 1e-5f) * gamma[c] + beta[c];
  X[idx] = (y > 0.f) ? y : 0.01f * y;
}

__global__ __launch_bounds__(THREADS)
void k_combine(const float* __restrict__ xg3, const float* __restrict__ xl3,
               float* __restrict__ out, int n) {
  int i = blockIdx.x * THREADS + threadIdx.x;
  if (i < n) out[i] = 0.01f * xg3[i] + 0.99f * xl3[i];
}

// ---------------------------------------------------------------------------
extern "C" void kernel_launch(void* const* d_in, const int* in_sizes, int n_in,
                              void* d_out, int out_size, void* d_ws, size_t ws_size,
                              hipStream_t stream) {
  (void)in_sizes; (void)n_in; (void)out_size; (void)ws_size;
  const float* z     = (const float*)d_in[0];
  const float* lin_w = (const float*)d_in[1];
  const float* lin_b = (const float*)d_in[2];
  const float* loc_w = (const float*)d_in[3];
  const float* loc_b = (const float*)d_in[4];
  const float* Wg    = (const float*)d_in[5];
  // d_in[6,7] = ug, cg: softmax over a single head == 1.0 -> unused.
  const float* bg    = (const float*)d_in[8];
  const float* Wg3   = (const float*)d_in[9];
  // d_in[10,11] = ug3, cg3: unused (single head).
  const float* bg3   = (const float*)d_in[12];
  const float* Wl    = (const float*)d_in[13];
  const float* bl    = (const float*)d_in[16];
  const float* Wl3   = (const float*)d_in[17];
  const float* bl3   = (const float*)d_in[20];
  const float* gam_g = (const float*)d_in[21];
  const float* bet_g = (const float*)d_in[22];
  const float* gam_l = (const float*)d_in[23];
  const float* bet_l = (const float*)d_in[24];
  const float* U0    = (const float*)d_in[25];
  const float* U1    = (const float*)d_in[26];
  const float* U2    = (const float*)d_in[27];
  const float* U3    = (const float*)d_in[28];
  const int*   A0    = (const int*)d_in[29];
  const int*   A1    = (const int*)d_in[30];
  const int*   A2    = (const int*)d_in[31];
  const int*   A3    = (const int*)d_in[32];
  float* out = (float*)d_out;

  // workspace layout (floats): ~9 MB
  float* ws    = (float*)d_ws;
  float* xbuf  = ws;                       // 256
  float* stats = ws + 256;                 // 64
  float* bufA  = ws + 1024;                // 16384*32
  float* bufB  = bufA + 16384 * 32;
  float* xw    = bufB + 16384 * 32;
  float* agg   = xw   + 16384 * 32;
  float* cnt   = agg  + 16384 * 32;        // 16384
  float* xg3   = cnt  + 16384;             // 16384*3
  float* xl3   = xg3  + 16384 * 3;         // 16384*3

  const int    Ns[4]  = {256, 1024, 4096, 16384};
  const int    Ms[4]  = {64, 256, 1024, 4096};
  const float* Ulv[4] = {U3, U2, U1, U0};
  const int*   Alv[4] = {A3, A2, A1, A0};

  auto zero = [&](float* p, int n) {
    k_zero<<<(n + THREADS - 1) / THREADS, THREADS, 0, stream>>>(p, n);
  };
  auto feast32 = [&](const float* Xin, const float* W, const float* b,
                     const int* edges, int n, float* Yout) {
    k_gemm_wmma<<<n / 128, THREADS, 0, stream>>>(Xin, W, xw, n, 32);
    zero(agg, n * 32);
    zero(cnt, n);
    int E = 6 * n;
    k_scatter<<<(E + n + THREADS - 1) / THREADS, THREADS, 0, stream>>>(
        edges, E, n, xw, agg, cnt, 32);
    k_feast_fin<<<(n * 32 + THREADS - 1) / THREADS, THREADS, 0, stream>>>(
        agg, cnt, b, Yout, n, 32);
  };
  auto feast3 = [&](const float* Xin, const float* W3, const float* b3,
                    const int* edges, int n, float* Yout) {
    k_xw3<<<(n + THREADS - 1) / THREADS, THREADS, 0, stream>>>(Xin, W3, xw, n);
    zero(agg, n * 3);
    zero(cnt, n);
    int E = 6 * n;
    k_scatter<<<(E + n + THREADS - 1) / THREADS, THREADS, 0, stream>>>(
        edges, E, n, xw, agg, cnt, 3);
    k_feast_fin<<<(n * 3 + THREADS - 1) / THREADS, THREADS, 0, stream>>>(
        agg, cnt, b3, Yout, n, 3);
  };
  auto bn = [&](float* X, const float* gamma, const float* beta, int n) {
    zero(stats, 64);
    k_bn_stats<<<128, THREADS, 0, stream>>>(X, stats, n);
    k_bn_apply<<<(n * 32 + THREADS - 1) / THREADS, THREADS, 0, stream>>>(
        X, stats, gamma, beta, n);
  };

  // ---------------- global branch ----------------
  k_lin<<<1, THREADS, 0, stream>>>(z, lin_w, lin_b, xbuf, bufA);
  float* cur = bufA;
  float* alt = bufB;
  for (int i = 0; i < 4; ++i) {
    const int N = Ns[i], M = Ms[i];
    k_gemm_wmma<<<N / 128, THREADS, 0, stream>>>(Ulv[i], cur, alt, N, M);
    { float* t = cur; cur = alt; alt = t; }
    if (i < 3) {
      feast32(cur, Wg + i * 1024, bg + i * 32, Alv[i], N, alt);
      { float* t = cur; cur = alt; alt = t; }
      bn(cur, gam_g + i * 32, bet_g + i * 32, N);
    } else {
      feast3(cur, Wg3, bg3, Alv[i], N, xg3);
    }
  }

  // ---------------- local branch (reuses bufA/bufB) ----------------
  k_loc<<<512, THREADS, 0, stream>>>(xbuf, loc_w, loc_b, bufA);
  cur = bufA;
  alt = bufB;
  for (int i = 0; i < 4; ++i) {
    if (i < 3) {
      feast32(cur, Wl + i * 1024, bl + i * 32, A0, 16384, alt);
      { float* t = cur; cur = alt; alt = t; }
      bn(cur, gam_l + i * 32, bet_l + i * 32, 16384);
    } else {
      feast3(cur, Wl3, bl3, A0, 16384, xl3);
    }
  }

  // ---------------- combine ----------------
  k_combine<<<(16384 * 3 + THREADS - 1) / THREADS, THREADS, 0, stream>>>(
      xg3, xl3, out, 16384 * 3);
}
